// MaskedMultiHeadAttention_39247411151188
// MI455X (gfx1250) — matrix operand order repaired and checked by reading, not compiled, not run on hardware
//
#include <hip/hip_runtime.h>
#include <hip/hip_bf16.h>
#include <math.h>

// ---------------------------------------------------------------------------
// MaskedMultiHeadAttention for MI455X (gfx1250, wave32, WMMA)
// B=2, S=2048, D=1024, H=16, HD=64
// LN -> f16 -> QKV WMMA GEMMs -> RoPE in place (Q pre-scaled by
// log2(e)/sqrt(HD)) -> V transpose -> flash attention (WMMA, exp2-domain
// softmax via raw v_exp_f32, mask-free fast path, rowsum-via-WMMA) ->
// output WMMA GEMM + residual (f32)
//
// Fragment layouts (ISA 7.12.2/7.12.4):
//   A 16x32 f16 : lane row M=lane%16; halves: K = 16g+h  (contiguous 16, same order as B)
//   B 32x16 f16 : lane col N=lane%16; halves: K = 16g+h  (contiguous 16; this code's order, not the ISA table order)
//   C/D 16x16   : lane col N=lane%16; vgpr r -> row M = r+8g
// ---------------------------------------------------------------------------

#define BB 2
#define SS 2048
#define DD 1024
#define HH 16
#define HD 64
#define NROWS (BB * SS)   // 4096

typedef __attribute__((ext_vector_type(16))) _Float16 v16h;
typedef __attribute__((ext_vector_type(8)))  float    v8f;

// raw v_exp_f32 (base-2); args are always <= 0 here, -inf -> 0
#define EXP2(x) __builtin_amdgcn_exp2f(x)

// ---------------------------------------------------------------------------
// 1) LayerNorm: x (f32 [4096,1024]) -> xn (f16)
// ---------------------------------------------------------------------------
__global__ void ln_kernel(const float* __restrict__ x,
                          const float* __restrict__ gamma,
                          const float* __restrict__ beta,
                          _Float16* __restrict__ xn) {
    const int row = blockIdx.x;
    const float* xr = x + (size_t)row * DD;
    __shared__ float ssum[256];
    __shared__ float ssq[256];
    float s = 0.f, s2 = 0.f;
    for (int i = threadIdx.x; i < DD; i += 256) {
        float v = xr[i];
        s += v; s2 += v * v;
    }
    ssum[threadIdx.x] = s; ssq[threadIdx.x] = s2;
    __syncthreads();
    for (int st = 128; st > 0; st >>= 1) {
        if (threadIdx.x < st) {
            ssum[threadIdx.x] += ssum[threadIdx.x + st];
            ssq[threadIdx.x]  += ssq[threadIdx.x + st];
        }
        __syncthreads();
    }
    const float mu  = ssum[0] * (1.0f / DD);
    const float var = ssq[0] * (1.0f / DD) - mu * mu;
    const float rstd = rsqrtf(var + 1e-5f);
    for (int i = threadIdx.x; i < DD; i += 256) {
        float v = (xr[i] - mu) * rstd * gamma[i] + beta[i];
        xn[(size_t)row * DD + i] = (_Float16)v;
    }
}

// ---------------------------------------------------------------------------
// 2) Weight convert f32->f16, layout preserved ([e][d] is K-contiguous)
// ---------------------------------------------------------------------------
__global__ void wcvt_kernel(const float* __restrict__ W,
                            _Float16* __restrict__ Wh) {
    const int i = (blockIdx.x * 256 + threadIdx.x) * 4;
    const float4 v = *(const float4*)(W + i);
    union { _Float16 h[4]; uint2 u; } o;
    o.h[0] = (_Float16)v.x; o.h[1] = (_Float16)v.y;
    o.h[2] = (_Float16)v.z; o.h[3] = (_Float16)v.w;
    *(uint2*)(Wh + i) = o.u;
}

// ---------------------------------------------------------------------------
// 3) WMMA GEMM: Y[m,n] = sum_k A[m,k] * W[n,k]
//    block = 256 thr = 8 waves, block tile 128(M) x 64(N), K step 32.
// ---------------------------------------------------------------------------
#define BPAD 40   // padded LDS row stride (halves)

template <int EPI>
__global__ void gemm_wmma(const _Float16* __restrict__ A,
                          const _Float16* __restrict__ Bw,   // [N=e][K=d]
                          _Float16* __restrict__ Yh,
                          float* __restrict__ Yf,
                          const float* __restrict__ resid) {
    __shared__ _Float16 ldsB[64 * BPAD];
    const int tid  = threadIdx.x;
    const int wave = tid >> 5;
    const int lane = tid & 31;
    const int g  = lane >> 4;
    const int ln = lane & 15;
    const int mb = blockIdx.x * 128 + wave * 16;
    const int nb = blockIdx.y * 64;
    const int row = mb + ln;

    v8f acc[4];
#pragma unroll
    for (int t = 0; t < 4; ++t)
#pragma unroll
        for (int r = 0; r < 8; ++r) acc[t][r] = 0.0f;

    const int br = tid >> 2;                  // 0..63 : e-row of B tile
    const int bc = (tid & 3) * 8;             // 0..24 : k chunk (16B)

    for (int kb = 0; kb < DD; kb += 32) {
        *(uint4*)(&ldsB[br * BPAD + bc]) =
            *(const uint4*)(Bw + (size_t)(nb + br) * DD + kb + bc);
        if (kb + 32 < DD)
            __builtin_prefetch(Bw + (size_t)(nb + br) * DD + kb + 32 + bc, 0, 0);
        __syncthreads();

        v16h a;
        {
            const _Float16* ap = A + (size_t)row * DD + kb + g * 16;
            ((uint4*)&a)[0] = *(const uint4*)(ap);
            ((uint4*)&a)[1] = *(const uint4*)(ap + 8);
        }

#pragma unroll
        for (int t = 0; t < 4; ++t) {
            v16h b;
            const _Float16* bp = &ldsB[(t * 16 + ln) * BPAD + g * 16];
            ((uint4*)&b)[0] = *(const uint4*)(bp);
            ((uint4*)&b)[1] = *(const uint4*)(bp + 8);
            acc[t] = __builtin_amdgcn_wmma_f32_16x16x32_f16(
                false, a, false, b, (short)0, acc[t], false, false);
        }
        __syncthreads();
    }

#pragma unroll
    for (int t = 0; t < 4; ++t) {
#pragma unroll
        for (int r = 0; r < 8; ++r) {
            const int m = mb + r + 8 * g;
            const int n = nb + t * 16 + ln;
            if (EPI == 0) {
                Yh[(size_t)m * DD + n] = (_Float16)acc[t][r];
            } else {
                Yf[(size_t)m * DD + n] = resid[(size_t)m * DD + n] + acc[t][r];
            }
        }
    }
}

// ---------------------------------------------------------------------------
// 4) RoPE: Q and K rotated in place. Q additionally pre-scaled by
//    log2(e)/sqrt(HD) so attention scores are directly in the exp2 domain.
// ---------------------------------------------------------------------------
#define QSCALE 0.1803368801111244f   // (1/8) * log2(e)

__global__ void rope_kernel(_Float16* __restrict__ Qp,
                            _Float16* __restrict__ Kp) {
    const int bs = blockIdx.x;
    const int s = bs % SS;
    for (int p = threadIdx.x; p < DD / 2; p += 256) {
        const int h = p >> 5;
        const int i = p & 31;
        const float inv = __powf(10000.0f, -(float)i / 32.0f);
        const float ang = (float)s * inv;
        const float c = __cosf(ang), sn = __sinf(ang);
        const float cq = c * QSCALE, snq = sn * QSCALE;
        const size_t base = (size_t)bs * DD + h * HD;
        const float q1 = (float)Qp[base + i], q2 = (float)Qp[base + i + 32];
        Qp[base + i]      = (_Float16)(q1 * cq - q2 * snq);
        Qp[base + i + 32] = (_Float16)(q2 * cq + q1 * snq);
        const float k1 = (float)Kp[base + i], k2 = (float)Kp[base + i + 32];
        Kp[base + i]      = (_Float16)(k1 * c - k2 * sn);
        Kp[base + i + 32] = (_Float16)(k2 * c + k1 * sn);
    }
}

// ---------------------------------------------------------------------------
// 5) V transpose per head: Vp[b,s,h*64+hd] -> Vt[((b*H+h)*HD+hd)*S + s]
// ---------------------------------------------------------------------------
__global__ void vtrans_kernel(const _Float16* __restrict__ Vp,
                              _Float16* __restrict__ Vt) {
    __shared__ _Float16 tile[32][33];
    const int sy  = blockIdx.y;
    const int bh  = sy >> 1;
    const int hd0 = (sy & 1) * 32;
    const int b = bh / HH, h = bh % HH;
    const int s0 = blockIdx.x * 32;
    const int tx = threadIdx.x, ty = threadIdx.y;
#pragma unroll
    for (int i = 0; i < 4; ++i)
        tile[ty + i * 8][tx] =
            Vp[(size_t)(b * SS + s0 + ty + i * 8) * DD + h * HD + hd0 + tx];
    __syncthreads();
#pragma unroll
    for (int i = 0; i < 4; ++i)
        Vt[((size_t)bh * HD + hd0 + ty + i * 8) * SS + s0 + tx] =
            tile[tx][ty + i * 8];
}

// ---------------------------------------------------------------------------
// 6) Flash attention step (32 keys). Scores arrive already in the exp2
//    domain (Q pre-scaled). Row max via shfl tree; row sum via WMMA with an
//    all-ones B operand (accL rescaled together with ctx accumulators).
// ---------------------------------------------------------------------------
template <bool MASKED>
__device__ __forceinline__ void attn_step(
    int k0, int q0, int g, int ln,
    const _Float16* __restrict__ kbase,   // Kp + b*S*D + h*HD
    const _Float16* __restrict__ vtb,     // Vt + bh*HD*S
    const v16h& aQ0, const v16h& aQ1, const v16h& bOnes,
    v8f (&acc)[4], v8f& accL, float (&m_i)[8],
    _Float16* __restrict__ PlW)           // per-wave 16x32 LDS tile
{
    // ---- scores ----
    v8f sc[2];
#pragma unroll
    for (int t = 0; t < 2; ++t) {
        const _Float16* kp = kbase + (size_t)(k0 + t * 16 + ln) * DD + g * 16;
        v16h bK0, bK1;
        ((uint4*)&bK0)[0] = *(const uint4*)(kp);
        ((uint4*)&bK0)[1] = *(const uint4*)(kp + 8);
        ((uint4*)&bK1)[0] = *(const uint4*)(kp + 32);
        ((uint4*)&bK1)[1] = *(const uint4*)(kp + 40);
        v8f z;
#pragma unroll
        for (int r = 0; r < 8; ++r) z[r] = 0.0f;
        z = __builtin_amdgcn_wmma_f32_16x16x32_f16(false, aQ0, false, bK0,
                                                   (short)0, z, false, false);
        z = __builtin_amdgcn_wmma_f32_16x16x32_f16(false, aQ1, false, bK1,
                                                   (short)0, z, false, false);
        if (MASKED) {
#pragma unroll
            for (int r = 0; r < 8; ++r) {
                const int q  = q0 + r + 8 * g;
                const int kk = k0 + t * 16 + ln;
                sc[t][r] = (kk <= q) ? z[r] : -INFINITY;
            }
        } else {
            sc[t] = z;
        }
    }

    // ---- online softmax in exp2 domain (raw v_exp_f32) ----
#pragma unroll
    for (int r = 0; r < 8; ++r) {
        float rm = fmaxf(sc[0][r], sc[1][r]);
#pragma unroll
        for (int off = 8; off > 0; off >>= 1)
            rm = fmaxf(rm, __shfl_xor(rm, off, 32));
        rm = fmaxf(m_i[r], rm);
        const float scale = EXP2(m_i[r] - rm);
        m_i[r] = rm;
        const float p0 = EXP2(sc[0][r] - rm);
        const float p1 = EXP2(sc[1][r] - rm);
#pragma unroll
        for (int t = 0; t < 4; ++t) acc[t][r] *= scale;
        accL[r] *= scale;
        const int M = r + 8 * g;
        PlW[M * 32 + ln]      = (_Float16)p0;
        PlW[M * 32 + 16 + ln] = (_Float16)p1;
    }

    // ---- reload P as A-fragment (same-wave DScnt ordering) ----
    v16h aP;
    {
        const _Float16* pl = PlW + ln * 32 + g * 16;
        ((uint4*)&aP)[0] = *(const uint4*)(pl);
        ((uint4*)&aP)[1] = *(const uint4*)(pl + 8);
    }

    // ---- row sums: accL += P x ones ----
    accL = __builtin_amdgcn_wmma_f32_16x16x32_f16(
        false, aP, false, bOnes, (short)0, accL, false, false);

    // ---- ctx += P x V ----
#pragma unroll
    for (int t = 0; t < 4; ++t) {
        const _Float16* vb = vtb + (size_t)(t * 16 + ln) * SS + k0 + g * 16;
        v16h bV;
        ((uint4*)&bV)[0] = *(const uint4*)(vb);
        ((uint4*)&bV)[1] = *(const uint4*)(vb + 8);
        acc[t] = __builtin_amdgcn_wmma_f32_16x16x32_f16(
            false, aP, false, bV, (short)0, acc[t], false, false);
    }
}

__global__ __launch_bounds__(128, 1)
void attn_kernel(const _Float16* __restrict__ Qp,
                 const _Float16* __restrict__ Kp,
                 const _Float16* __restrict__ Vt,
                 _Float16* __restrict__ Ctx) {
    __shared__ _Float16 Pl[4][16 * 32];
    const int wave = threadIdx.x >> 5;
    const int lane = threadIdx.x & 31;
    const int g  = lane >> 4;
    const int ln = lane & 15;
    const int bh = blockIdx.y;
    const int b = bh / HH, h = bh % HH;
    const int q0 = (blockIdx.x * 4 + wave) * 16;

    v16h aQ0, aQ1;
    {
        const _Float16* qb = Qp + ((size_t)(b * SS + q0 + ln) * DD) + h * HD;
        ((uint4*)&aQ0)[0] = *(const uint4*)(qb + g * 16);
        ((uint4*)&aQ0)[1] = *(const uint4*)(qb + 8 + g * 16);
        ((uint4*)&aQ1)[0] = *(const uint4*)(qb + 32 + g * 16);
        ((uint4*)&aQ1)[1] = *(const uint4*)(qb + 40 + g * 16);
    }
    v16h bOnes;
#pragma unroll
    for (int hh = 0; hh < 16; ++hh) bOnes[hh] = (_Float16)1.0f;

    v8f acc[4], accL;
#pragma unroll
    for (int t = 0; t < 4; ++t)
#pragma unroll
        for (int r = 0; r < 8; ++r) acc[t][r] = 0.0f;
#pragma unroll
    for (int r = 0; r < 8; ++r) accL[r] = 0.0f;
    float m_i[8];
#pragma unroll
    for (int r = 0; r < 8; ++r) m_i[r] = -INFINITY;

    const _Float16* kbase = Kp + (size_t)b * SS * DD + h * HD;
    const _Float16* vtb   = Vt + (size_t)bh * HD * SS;
    _Float16* PlW = &Pl[wave][0];

    // A 32-key tile is fully unmasked iff k0+31 <= q0.
    const int kfull = (q0 >= 31) ? ((((q0 - 31) >> 5) + 1) << 5) : 0;
    const int kend  = q0 + 15;

    for (int k0 = 0; k0 < kfull; k0 += 32)
        attn_step<false>(k0, q0, g, ln, kbase, vtb, aQ0, aQ1, bOnes,
                         acc, accL, m_i, PlW);
    for (int k0 = kfull; k0 <= kend; k0 += 32)
        attn_step<true>(k0, q0, g, ln, kbase, vtb, aQ0, aQ1, bOnes,
                        acc, accL, m_i, PlW);

    // ---- normalize (accL holds row sums in every lane of the group) ----
#pragma unroll
    for (int t = 0; t < 4; ++t) {
#pragma unroll
        for (int r = 0; r < 8; ++r) {
            const int srow = q0 + r + 8 * g;
            Ctx[(size_t)(b * SS + srow) * DD + h * HD + t * 16 + ln] =
                (_Float16)(acc[t][r] / accL[r]);
        }
    }
}

// ---------------------------------------------------------------------------
// Launcher. Workspace (~48 MB):
//   xn/Ctx (aliased) @0        (8 MB)
//   Wq/Wk/Wv/Wo f16  @8M+w*2M  (8 MB)
//   Qp @16M  Kp @24M  Vp @32M  (8 MB each)
//   Vt @40M                    (8 MB)
// ---------------------------------------------------------------------------
extern "C" void kernel_launch(void* const* d_in, const int* in_sizes, int n_in,
                              void* d_out, int out_size, void* d_ws, size_t ws_size,
                              hipStream_t stream) {
    const float* x     = (const float*)d_in[0];
    const float* W_q   = (const float*)d_in[1];
    const float* W_k   = (const float*)d_in[2];
    const float* W_v   = (const float*)d_in[3];
    const float* W_o   = (const float*)d_in[4];
    const float* gamma = (const float*)d_in[5];
    const float* beta  = (const float*)d_in[6];
    float* out = (float*)d_out;

    char* ws = (char*)d_ws;
    const size_t MB = 1024 * 1024;
    _Float16* xn  = (_Float16*)(ws + 0);
    _Float16* Ctx = xn;                    // xn dead after QKV GEMMs
    _Float16* Wqh = (_Float16*)(ws + 8 * MB);
    _Float16* Wkh = (_Float16*)(ws + 10 * MB);
    _Float16* Wvh = (_Float16*)(ws + 12 * MB);
    _Float16* Woh = (_Float16*)(ws + 14 * MB);
    _Float16* Qp = (_Float16*)(ws + 16 * MB);
    _Float16* Kp = (_Float16*)(ws + 24 * MB);
    _Float16* Vp = (_Float16*)(ws + 32 * MB);
    _Float16* Vt = (_Float16*)(ws + 40 * MB);

    ln_kernel<<<NROWS, 256, 0, stream>>>(x, gamma, beta, xn);

    const int wcg = (DD * DD) / (256 * 4);
    wcvt_kernel<<<wcg, 256, 0, stream>>>(W_q, Wqh);
    wcvt_kernel<<<wcg, 256, 0, stream>>>(W_k, Wkh);
    wcvt_kernel<<<wcg, 256, 0, stream>>>(W_v, Wvh);
    wcvt_kernel<<<wcg, 256, 0, stream>>>(W_o, Woh);

    dim3 gg(NROWS / 128, DD / 64);
    gemm_wmma<0><<<gg, 256, 0, stream>>>(xn, Wqh, Qp, nullptr, nullptr);
    gemm_wmma<0><<<gg, 256, 0, stream>>>(xn, Wkh, Kp, nullptr, nullptr);
    gemm_wmma<0><<<gg, 256, 0, stream>>>(xn, Wvh, Vp, nullptr, nullptr);

    rope_kernel<<<NROWS, 256, 0, stream>>>(Qp, Kp);

    vtrans_kernel<<<dim3(SS / 32, BB * HH * 2), dim3(32, 8), 0, stream>>>(Vp, Vt);

    attn_kernel<<<dim3(SS / 64, BB * HH), 128, 0, stream>>>(Qp, Kp, Vt, Ctx);

    gemm_wmma<1><<<gg, 256, 0, stream>>>(Ctx, Woh, nullptr, out, x);
}
